// GATLayer_51814485459561
// MI455X (gfx1250) — matrix operand order repaired and checked by reading, not compiled, not run on hardware
//
#include <hip/hip_runtime.h>
#include <hip/hip_bf16.h>
#include <math.h>

// ---------------------------------------------------------------------------
// Types for CDNA5 WMMA (wave32): A/B = 16 bf16 (8 VGPRs), C/D = 8 f32.
// ---------------------------------------------------------------------------
typedef __attribute__((ext_vector_type(16))) __bf16 v16bf;
typedef __attribute__((ext_vector_type(8)))  float  v8f;
typedef __attribute__((ext_vector_type(4)))  int    v4i;

union BF16x16 { v16bf v; unsigned short u[16]; uint4 q[2]; };
union F32x8   { v8f   v; float f[8]; };

// Optional gfx1250 async global->LDS path (compile-safe if builtins absent).
#if __has_builtin(__builtin_amdgcn_global_load_async_to_lds_b128) && \
    __has_builtin(__builtin_amdgcn_s_wait_asynccnt)
#define HAVE_ASYNC_LDS 1
typedef __attribute__((address_space(1))) v4i* gv4_ptr;   // global b128
typedef __attribute__((address_space(3))) v4i* lv4_ptr;   // LDS b128
#else
#define HAVE_ASYNC_LDS 0
#endif

__device__ __forceinline__ unsigned short f2bf(float f) {
  union { float f; unsigned u; } cv; cv.f = f;
  unsigned u = cv.u;
  unsigned r = u + 0x7fffu + ((u >> 16) & 1u);   // round-to-nearest-even
  return (unsigned short)(r >> 16);
}

// Monotone order-preserving mapping float -> u32 so atomicMax(u32) == float max.
__device__ __forceinline__ unsigned fkey(float f) {
  unsigned b = __float_as_uint(f);
  return (b & 0x80000000u) ? ~b : (b | 0x80000000u);
}
__device__ __forceinline__ float fdec(unsigned k) {
  if (k == 0u) return 0.0f;                       // empty segment -> 0 (ref semantics)
  return (k & 0x80000000u) ? __uint_as_float(k & 0x7fffffffu)
                           : __uint_as_float(~k);
}

__device__ __forceinline__ float lrelu(float e) { return e > 0.f ? e : 0.2f * e; }

// ---------------------------------------------------------------------------
// Layer-1 GEMM: h1[N,128] = x[N,128] @ W1[128,128], bf16 WMMA, f32 accum.
// Block = 256 threads = 8 waves; block owns a 16-row stripe, wave w owns the
// 16x16 output tile at columns w*16..w*16+15.  K=128 -> 4 WMMA ops/wave.
// LDS: A as bf16 [16 x 128] (pitch 136), W transposed bf16 Bt[n][k]
// (pitch 136) so each lane's fragment is two aligned 16B runs -> ds_load_b128.
// ---------------------------------------------------------------------------
#define LDP 136  // LDS row pitch (bf16 elems); 272B rows -> conflict-free b128

__global__ __launch_bounds__(256) void gemm1_wmma_kernel(
    const float* __restrict__ x, const float* __restrict__ W,
    float* __restrict__ h1, int N) {
  __shared__ __align__(16) unsigned short As[16 * LDP];    // A tile (bf16)
  __shared__ __align__(16) unsigned short Bt[128 * LDP];   // W^T: [n][k] (bf16)
#if HAVE_ASYNC_LDS
  __shared__ __align__(16) float Xst[16 * 128];            // async f32 stage
#endif
  const int tid = threadIdx.x;
  const int mBase = blockIdx.x * 16;
  const bool fullTile = (mBase + 16 <= N);

#if HAVE_ASYNC_LDS
  // Stage the x tile straight into LDS with async b128 copies (no VGPR trip).
  for (int i = tid; i < 16 * 32; i += 256) {               // float4 granules
    int r = i >> 5, c4 = (i & 31) * 4;
    int row = mBase + r;
    int srow = (row < N) ? row : (N - 1);                  // clamp; masked later
    const float* srcp = x + (size_t)srow * 128 + c4;
    __builtin_amdgcn_global_load_async_to_lds_b128(
        (gv4_ptr)(size_t)srcp, (lv4_ptr)(void*)&Xst[r * 128 + c4], 0, 0);
  }
  __builtin_amdgcn_s_wait_asynccnt(0);
  __syncthreads();
#endif

  // Convert A tile to bf16 (ushort4 -> ds_store_b64).
  for (int i = tid; i < 16 * 32; i += 256) {
    int r = i >> 5, c4 = (i & 31) * 4;
    int row = mBase + r;
    float4 v = {0.f, 0.f, 0.f, 0.f};
    if (fullTile || row < N) {
#if HAVE_ASYNC_LDS
      v = *(const float4*)&Xst[r * 128 + c4];
#else
      v = *(const float4*)&x[(size_t)row * 128 + c4];
#endif
    }
    ushort4 o;
    o.x = f2bf(v.x); o.y = f2bf(v.y); o.z = f2bf(v.z); o.w = f2bf(v.w);
    *(ushort4*)&As[r * LDP + c4] = o;
  }
  // Load W row-major (coalesced float4), store transposed bf16 into Bt[n][k].
  for (int i = tid; i < 128 * 32; i += 256) {
    int k = i >> 5, n4 = (i & 31) * 4;
    float4 v = *(const float4*)&W[(size_t)k * 128 + n4];
    Bt[(n4 + 0) * LDP + k] = f2bf(v.x);
    Bt[(n4 + 1) * LDP + k] = f2bf(v.y);
    Bt[(n4 + 2) * LDP + k] = f2bf(v.z);
    Bt[(n4 + 3) * LDP + k] = f2bf(v.w);
  }
  __syncthreads();

  const int wave  = tid >> 5;
  const int lane  = tid & 31;
  const int half  = lane >> 4;               // 0: lanes 0-15, 1: lanes 16-31
  const int ml    = lane & 15;               // A row / B col within tile
  const int nBase = wave * 16;
  const int aRow  = ml * LDP;
  const int bRow  = (nBase + ml) * LDP;

  F32x8 c;
#pragma unroll
  for (int r = 0; r < 8; ++r) c.f[r] = 0.f;

#pragma unroll
  for (int kk = 0; kk < 4; ++kk) {
    const int k0 = kk * 32;
    BF16x16 a, b;
    // A lane fragment: K = k0+8*half+{0..7} and k0+16+8*half+{0..7}
    a.q[0] = *(const uint4*)&As[aRow + k0 + 8 * half];
    a.q[1] = *(const uint4*)&As[aRow + k0 + 16 + 8 * half];
    // B lane fragment (col nBase+ml): K = k0+8*half+{0..7} and k0+16+8*half+{0..7}
    b.q[0] = *(const uint4*)&Bt[bRow + k0 + 8 * half];
    b.q[1] = *(const uint4*)&Bt[bRow + k0 + 16 + 8 * half];
    c.v = __builtin_amdgcn_wmma_f32_16x16x32_bf16(
        false, a.v, false, b.v, (short)0, c.v, false, false);
  }

  // C/D layout: lane column = ml, VGPR r -> row r + 8*half.
  // Uniform fast path: one base address + 8 immediate-offset stores.
  if (fullTile) {
    float* outp = h1 + (size_t)(mBase + 8 * half) * 128 + nBase + ml;
#pragma unroll
    for (int r = 0; r < 8; ++r) outp[(size_t)r * 128] = c.f[r];
  } else {
#pragma unroll
    for (int r = 0; r < 8; ++r) {
      int row = mBase + r + 8 * half;
      if (row < N) h1[(size_t)row * 128 + nBase + ml] = c.f[r];
    }
  }
}

// ---------------------------------------------------------------------------
// Attention scores per node/head: s_src[n,h] = h1[n,h,:]·a_src[h,:], same dst.
// ---------------------------------------------------------------------------
__global__ void scores1_kernel(const float* __restrict__ h1,
                               const float* __restrict__ a_src,
                               const float* __restrict__ a_dst,
                               float* __restrict__ ss, float* __restrict__ sd,
                               int N) {
  int t = blockIdx.x * blockDim.x + threadIdx.x;
  if (t >= N * 4) return;
  int n = t >> 2, h = t & 3;
  const float* hp = h1 + (size_t)n * 128 + h * 32;
  const float* as = a_src + h * 32;
  const float* ad = a_dst + h * 32;
  float s0 = 0.f, s1 = 0.f;
#pragma unroll 8
  for (int c = 0; c < 32; ++c) { float v = hp[c]; s0 += v * as[c]; s1 += v * ad[c]; }
  ss[t] = s0; sd[t] = s1;
}

// ---------------------------------------------------------------------------
// Edge passes (templated over heads H).  Virtual edge idx >= E are self-loops.
// ---------------------------------------------------------------------------
template <int H>
__global__ void edge_max_kernel(const int* __restrict__ src, const int* __restrict__ dst,
                                int E, int N,
                                const float* __restrict__ ss, const float* __restrict__ sd,
                                unsigned* __restrict__ mkey) {
  int idx = blockIdx.x * blockDim.x + threadIdx.x;
  if (idx >= E + N) return;
  int s, d;
  if (idx < E) { s = src[idx]; d = dst[idx]; } else { s = d = idx - E; }
#pragma unroll
  for (int h = 0; h < H; ++h) {
    float e = lrelu(ss[s * H + h] + sd[d * H + h]);
    atomicMax(&mkey[d * H + h], fkey(e));
  }
}

template <int H>
__global__ void edge_den_kernel(const int* __restrict__ src, const int* __restrict__ dst,
                                int E, int N,
                                const float* __restrict__ ss, const float* __restrict__ sd,
                                const unsigned* __restrict__ mkey,
                                float* __restrict__ den) {
  int idx = blockIdx.x * blockDim.x + threadIdx.x;
  if (idx >= E + N) return;
  int s, d;
  if (idx < E) { s = src[idx]; d = dst[idx]; } else { s = d = idx - E; }
#pragma unroll
  for (int h = 0; h < H; ++h) {
    float e  = lrelu(ss[s * H + h] + sd[d * H + h]);
    float ex = expf(e - fdec(mkey[d * H + h]));
    atomicAdd(&den[d * H + h], ex);
  }
}

// Layer-1 aggregation: 2 edges / block, 128 ch / edge (thread c -> channel c).
__global__ __launch_bounds__(256) void edge_agg1_kernel(
    const int* __restrict__ src, const int* __restrict__ dst, int E, int N,
    const float* __restrict__ ss, const float* __restrict__ sd,
    const unsigned* __restrict__ mkey, const float* __restrict__ den,
    const float* __restrict__ h1, float* __restrict__ agg) {
  int t = threadIdx.x;
  int c = t & 127;
  long long idx = (long long)blockIdx.x * 2 + (t >> 7);
  if (idx >= (long long)E + N) return;
  int s, d;
  if (idx < E) { s = src[idx]; d = dst[idx]; } else { s = d = (int)(idx - E); }
  if (c == 0) __builtin_prefetch(h1 + (size_t)s * 128, 0, 0);  // global_prefetch_b8
  int h = c >> 5;
  float e     = lrelu(ss[s * 4 + h] + sd[d * 4 + h]);
  float alpha = expf(e - fdec(mkey[d * 4 + h])) / (den[d * 4 + h] + 1e-16f);
  atomicAdd(&agg[(size_t)d * 128 + c], alpha * h1[(size_t)s * 128 + c]);
}

// ---------------------------------------------------------------------------
// BatchNorm statistics (population), LDS ds_add_f32 reduce -> global atomics.
// ---------------------------------------------------------------------------
__global__ void bn_stats_kernel(const float* __restrict__ agg, const float* __restrict__ b1,
                                float* __restrict__ gsum, float* __restrict__ gsq,
                                long long total) {
  __shared__ float ssum[128], ssq[128];
  int tid = threadIdx.x;
  if (tid < 128) { ssum[tid] = 0.f; ssq[tid] = 0.f; }
  __syncthreads();
  long long stride = (long long)gridDim.x * blockDim.x;
  for (long long i = (long long)blockIdx.x * blockDim.x + tid; i < total; i += stride) {
    int c = (int)(i & 127);
    float y = agg[i] + b1[c];
    atomicAdd(&ssum[c], y);
    atomicAdd(&ssq[c], y * y);
  }
  __syncthreads();
  if (tid < 128) { atomicAdd(&gsum[tid], ssum[tid]); atomicAdd(&gsq[tid], ssq[tid]); }
}

__global__ void bn_finalize_kernel(const float* __restrict__ gsum, const float* __restrict__ gsq,
                                   const float* __restrict__ gamma, const float* __restrict__ beta,
                                   int N, float* __restrict__ scale, float* __restrict__ shift) {
  int c = threadIdx.x;
  if (c >= 128) return;
  float invN = 1.0f / (float)N;
  float mu  = gsum[c] * invN;
  float var = gsq[c] * invN - mu * mu;
  float sc  = gamma[c] * rsqrtf(var + 1e-5f);
  scale[c] = sc;
  shift[c] = beta[c] - mu * sc;
}

__global__ void bn_apply_kernel(const float* __restrict__ agg, const float* __restrict__ b1,
                                const float* __restrict__ scale, const float* __restrict__ shift,
                                float* __restrict__ hbn, long long total) {
  long long i = (long long)blockIdx.x * blockDim.x + threadIdx.x;
  if (i >= total) return;
  int c = (int)(i & 127);
  float v = (agg[i] + b1[c]) * scale[c] + shift[c];
  hbn[i] = v > 0.f ? v : expm1f(v);   // ELU
}

// ---------------------------------------------------------------------------
// Layer-2 GEMM (Fout=2) + scores: tiny, VALU dot per node row.
// ---------------------------------------------------------------------------
__global__ void gemm2_scores_kernel(const float* __restrict__ hbn, const float* __restrict__ W2,
                                    const float* __restrict__ as2, const float* __restrict__ ad2,
                                    float* __restrict__ h2, float* __restrict__ s2s,
                                    float* __restrict__ s2d, int N) {
  int n = blockIdx.x * blockDim.x + threadIdx.x;
  if (n >= N) return;
  const float* row = hbn + (size_t)n * 128;
  float a0 = 0.f, a1 = 0.f;
#pragma unroll 8
  for (int k = 0; k < 128; ++k) {
    float v = row[k];
    a0 += v * W2[2 * k];
    a1 += v * W2[2 * k + 1];
  }
  h2[(size_t)n * 2]     = a0;
  h2[(size_t)n * 2 + 1] = a1;
  s2s[n] = a0 * as2[0] + a1 * as2[1];
  s2d[n] = a0 * ad2[0] + a1 * ad2[1];
}

// Layer-2 aggregation (H=1, C=2) directly into d_out (pre-seeded with b2).
__global__ void edge_agg2_kernel(const int* __restrict__ src, const int* __restrict__ dst,
                                 int E, int N,
                                 const float* __restrict__ ss, const float* __restrict__ sd,
                                 const unsigned* __restrict__ mkey, const float* __restrict__ den,
                                 const float* __restrict__ h2, float* __restrict__ out) {
  int idx = blockIdx.x * blockDim.x + threadIdx.x;
  if (idx >= E + N) return;
  int s, d;
  if (idx < E) { s = src[idx]; d = dst[idx]; } else { s = d = idx - E; }
  float e     = lrelu(ss[s] + sd[d]);
  float alpha = expf(e - fdec(mkey[d])) / (den[d] + 1e-16f);
  atomicAdd(&out[(size_t)d * 2],     alpha * h2[(size_t)s * 2]);
  atomicAdd(&out[(size_t)d * 2 + 1], alpha * h2[(size_t)s * 2 + 1]);
}

__global__ void init_out_kernel(float* __restrict__ out, const float* __restrict__ b2, int N) {
  int t = blockIdx.x * blockDim.x + threadIdx.x;
  if (t < N * 2) out[t] = b2[t & 1];
}

// ---------------------------------------------------------------------------
// Host-side orchestration (graph-capture safe: only async launches/memset).
// ---------------------------------------------------------------------------
extern "C" void kernel_launch(void* const* d_in, const int* in_sizes, int n_in,
                              void* d_out, int out_size, void* d_ws, size_t ws_size,
                              hipStream_t stream) {
  const float* x     = (const float*)d_in[0];
  const int*   ei    = (const int*)d_in[1];
  const float* W1    = (const float*)d_in[2];
  const float* as1   = (const float*)d_in[3];
  const float* ad1   = (const float*)d_in[4];
  const float* b1    = (const float*)d_in[5];
  const float* gamma = (const float*)d_in[6];
  const float* beta  = (const float*)d_in[7];
  const float* W2    = (const float*)d_in[8];
  const float* as2   = (const float*)d_in[9];
  const float* ad2   = (const float*)d_in[10];
  const float* b2    = (const float*)d_in[11];
  float* out = (float*)d_out;

  const int N = in_sizes[0] / 128;
  const int E = in_sizes[1] / 2;
  const int Etot = E + N;
  const int* src = ei;
  const int* dst = ei + E;

  // Workspace layout (floats).
  float* ws = (float*)d_ws;
  size_t sN = (size_t)N;
  float*    h1    = ws;                          // N*128 (reused as hbn)
  float*    ss1   = h1 + sN * 128;               // N*4
  float*    sd1   = ss1 + sN * 4;                // N*4
  unsigned* mkey1 = (unsigned*)(sd1 + sN * 4);   // N*4   } zero region start
  float*    den1  = (float*)(mkey1 + sN * 4);    // N*4
  float*    agg1  = den1 + sN * 4;               // N*128
  float*    gsum  = agg1 + sN * 128;             // 128
  float*    gsq   = gsum + 128;                  // 128
  unsigned* mkey2 = (unsigned*)(gsq + 128);      // N
  float*    den2  = (float*)(mkey2 + sN);        // N     } zero region end
  float*    bnsc  = den2 + sN;                   // 128
  float*    bnsh  = bnsc + 128;                  // 128
  float*    h2    = bnsh + 128;                  // N*2
  float*    s2s   = h2 + sN * 2;                 // N
  float*    s2d   = s2s + sN;                    // N

  size_t zeroBytes = ((size_t)138 * N + 256) * sizeof(float);
  (void)hipMemsetAsync((void*)mkey1, 0, zeroBytes, stream);

  init_out_kernel<<<(2 * N + 255) / 256, 256, 0, stream>>>(out, b2, N);

  // Layer 1
  gemm1_wmma_kernel<<<(N + 15) / 16, 256, 0, stream>>>(x, W1, h1, N);
  scores1_kernel<<<(4 * N + 255) / 256, 256, 0, stream>>>(h1, as1, ad1, ss1, sd1, N);
  edge_max_kernel<4><<<(Etot + 255) / 256, 256, 0, stream>>>(src, dst, E, N, ss1, sd1, mkey1);
  edge_den_kernel<4><<<(Etot + 255) / 256, 256, 0, stream>>>(src, dst, E, N, ss1, sd1, mkey1, den1);
  edge_agg1_kernel<<<(Etot + 1) / 2, 256, 0, stream>>>(src, dst, E, N, ss1, sd1, mkey1, den1, h1, agg1);

  // BatchNorm + ELU
  long long total = (long long)N * 128;
  bn_stats_kernel<<<1024, 256, 0, stream>>>(agg1, b1, gsum, gsq, total);
  bn_finalize_kernel<<<1, 128, 0, stream>>>(gsum, gsq, gamma, beta, N, bnsc, bnsh);
  bn_apply_kernel<<<(unsigned)((total + 255) / 256), 256, 0, stream>>>(agg1, b1, bnsc, bnsh, h1, total);

  // Layer 2
  gemm2_scores_kernel<<<(N + 255) / 256, 256, 0, stream>>>(h1, W2, as2, ad2, h2, s2s, s2d, N);
  edge_max_kernel<1><<<(Etot + 255) / 256, 256, 0, stream>>>(src, dst, E, N, s2s, s2d, mkey2);
  edge_den_kernel<1><<<(Etot + 255) / 256, 256, 0, stream>>>(src, dst, E, N, s2s, s2d, mkey2, den2);
  edge_agg2_kernel<<<(Etot + 255) / 256, 256, 0, stream>>>(src, dst, E, N, s2s, s2d, mkey2, den2, h2, out);
}
